// Block_42872363549182
// MI455X (gfx1250) — matrix operand order repaired and checked by reading, not compiled, not run on hardware
//
#include <hip/hip_runtime.h>
#include <hip/hip_bf16.h>
#include <math.h>

typedef _Float16 v16h  __attribute__((ext_vector_type(16)));
typedef _Float16 v8h   __attribute__((ext_vector_type(8)));
typedef _Float16 v4h   __attribute__((ext_vector_type(4)));
typedef float    v8f   __attribute__((ext_vector_type(8)));
typedef float    f32x4 __attribute__((ext_vector_type(4)));
typedef unsigned int u32x4 __attribute__((ext_vector_type(4)));
typedef int      iv4   __attribute__((ext_vector_type(4)));

// Address-space-qualified int4 pointers for the gfx1250 async-copy builtins
typedef __attribute__((address_space(1))) iv4 giv4;
typedef __attribute__((address_space(3))) iv4 liv4;

union AFrag { v16h v; v8h h8[2]; _Float16 h[16]; };

#define WMMA_F32_F16(a, b, c) \
  __builtin_amdgcn_wmma_f32_16x16x32_f16(false, (a), false, (b), (short)0, (c), false, false)

// ---------------------------------------------------------------------------
// CDNA5 async global->LDS copies (ASYNCcnt path), guarded so we always compile
// ---------------------------------------------------------------------------
#if __has_builtin(__builtin_amdgcn_global_load_async_to_lds_b128) && \
    __has_builtin(__builtin_amdgcn_s_wait_asynccnt)
#define HAVE_ASYNC 1
#else
#define HAVE_ASYNC 0
#endif

#if HAVE_ASYNC
#define WAIT_ASYNC(n) __builtin_amdgcn_s_wait_asynccnt(n)
#else
#define WAIT_ASYNC(n)
#endif

__device__ __forceinline__ void copy16_g2l(const _Float16* g, _Float16* l) {
#if HAVE_ASYNC
  __builtin_amdgcn_global_load_async_to_lds_b128((giv4*)g, (liv4*)l, 0, 0);
#else
  *(u32x4*)l = *(const u32x4*)g;
#endif
}

// ---------------------------------------------------------------------------
// 16-lane-row butterfly reductions via v_permlane16_b32 (VALU, no LDS traffic)
// ---------------------------------------------------------------------------
#if __has_builtin(__builtin_amdgcn_permlane16)
#define HAVE_PERMLANE16 1
#else
#define HAVE_PERMLANE16 0
#endif

template <int M, unsigned S0, unsigned S1>
__device__ __forceinline__ float bfly16(float x) {
#if HAVE_PERMLANE16
  unsigned u = __float_as_uint(x);
  return __uint_as_float(__builtin_amdgcn_permlane16(u, u, S0, S1, false, false));
#else
  return __shfl_xor(x, M);
#endif
}

__device__ __forceinline__ float rowmax16(float x) {
  x = fmaxf(x, bfly16<1, 0x67452301u, 0xEFCDAB89u>(x));
  x = fmaxf(x, bfly16<2, 0x54761032u, 0xDCFE98BAu>(x));
  x = fmaxf(x, bfly16<4, 0x32107654u, 0xBA98FEDCu>(x));
  x = fmaxf(x, bfly16<8, 0xFEDCBA98u, 0x76543210u>(x));
  return x;
}

__device__ __forceinline__ float rowsum16(float x) {
  x += bfly16<1, 0x67452301u, 0xEFCDAB89u>(x);
  x += bfly16<2, 0x54761032u, 0xDCFE98BAu>(x);
  x += bfly16<4, 0x32107654u, 0xBA98FEDCu>(x);
  x += bfly16<8, 0xFEDCBA98u, 0x76543210u>(x);
  return x;
}

// ---------------------------------------------------------------------------
// fp32 -> fp16 weight conversion (weights are fresh inputs every call)
// ---------------------------------------------------------------------------
__global__ void cvt_f32_f16(const float* __restrict__ in, _Float16* __restrict__ out, int n4) {
  int i = blockIdx.x * blockDim.x + threadIdx.x;
  if (i < n4) {
    f32x4 v = ((const f32x4*)in)[i];
    v4h o;
    o[0] = (_Float16)v[0]; o[1] = (_Float16)v[1];
    o[2] = (_Float16)v[2]; o[3] = (_Float16)v[3];
    ((v4h*)out)[i] = o;
  }
}

// ---------------------------------------------------------------------------
// LayerNorm over last dim (1024), one block per row, writes f16
// ---------------------------------------------------------------------------
__global__ __launch_bounds__(256) void ln_kernel(const float* __restrict__ x,
                                                 const float* __restrict__ w,
                                                 const float* __restrict__ b,
                                                 _Float16* __restrict__ out) {
  const int row = blockIdx.x;
  const int tid = threadIdx.x;
  const float* xr = x + (size_t)row * 1024;
  f32x4 v = *(const f32x4*)(xr + tid * 4);
  float s = v[0] + v[1] + v[2] + v[3];
  float q = v[0]*v[0] + v[1]*v[1] + v[2]*v[2] + v[3]*v[3];
  #pragma unroll
  for (int m = 1; m < 32; m <<= 1) { s += __shfl_xor(s, m); q += __shfl_xor(q, m); }
  __shared__ float ss[8], sq[8];
  if ((tid & 31) == 0) { ss[tid >> 5] = s; sq[tid >> 5] = q; }
  __syncthreads();
  float ts = 0.f, tq = 0.f;
  #pragma unroll
  for (int i = 0; i < 8; i++) { ts += ss[i]; tq += sq[i]; }
  float mu = ts * (1.0f / 1024.0f);
  float var = tq * (1.0f / 1024.0f) - mu * mu;
  float rs = rsqrtf(var + 1e-5f);
  f32x4 wv = *(const f32x4*)(w + tid * 4);
  f32x4 bv = *(const f32x4*)(b + tid * 4);
  v4h o;
  #pragma unroll
  for (int j = 0; j < 4; j++) o[j] = (_Float16)((v[j] - mu) * rs * wv[j] + bv[j]);
  *(v4h*)(out + (size_t)row * 1024 + tid * 4) = o;
}

// ---------------------------------------------------------------------------
// Tiled WMMA GEMM:  C[M x N] = A[M x K] (f16 row-major) * B[N x K]^T (weights)
// Block: 256 threads = 8 waves; tile 128x128, BK=32, double-buffered LDS with
// async global->LDS copies overlapping the WMMA work.
// EPI 0: QKV scatter  EPI 1: out = resid + gamma*(acc+bias)  EPI 2: GELU->f16
// ---------------------------------------------------------------------------
template <int EPI>
__global__ __launch_bounds__(256) void gemm128(
    const _Float16* __restrict__ A, const _Float16* __restrict__ Bw,
    const float* __restrict__ bias, const float* __restrict__ gamma,
    const float* __restrict__ resid, float* __restrict__ outF,
    _Float16* __restrict__ outH,
    _Float16* __restrict__ qout, _Float16* __restrict__ kout, _Float16* __restrict__ vout,
    int N, int K) {
  __shared__ __align__(16) _Float16 As[2][128 * 40];
  __shared__ __align__(16) _Float16 Bs[2][128 * 40];
  const int tid = threadIdx.x, lane = tid & 31, wave = tid >> 5;
  const int wm = wave & 3, wn = wave >> 2;
  const int bR = blockIdx.y * 128, bC = blockIdx.x * 128;
  const int lhalf = lane >> 4, l15 = lane & 15;

  v8f acc[2][4] = {};

  auto issue_tile = [&](int buf, int k0) {
    #pragma unroll
    for (int i = 0; i < 2; i++) {
      int idx = tid + i * 256;
      int r = idx >> 2, c = (idx & 3) * 8;
      copy16_g2l(&A[(size_t)(bR + r) * K + k0 + c], &As[buf][r * 40 + c]);
      copy16_g2l(&Bw[(size_t)(bC + r) * K + k0 + c], &Bs[buf][r * 40 + c]);
    }
  };

  auto compute_tile = [&](int buf) {
    AFrag af[2], bf[4];
    #pragma unroll
    for (int mt = 0; mt < 2; mt++) {
      const _Float16* p = &As[buf][(wm * 32 + mt * 16 + l15) * 40 + lhalf * 8];
      af[mt].h8[0] = *(const v8h*)p;
      af[mt].h8[1] = *(const v8h*)(p + 16);
    }
    #pragma unroll
    for (int nt = 0; nt < 4; nt++) {
      const _Float16* p = &Bs[buf][(wn * 64 + nt * 16 + l15) * 40 + lhalf * 8];
      bf[nt].h8[0] = *(const v8h*)p;
      bf[nt].h8[1] = *(const v8h*)(p + 16);
    }
    #pragma unroll
    for (int mt = 0; mt < 2; mt++)
      #pragma unroll
      for (int nt = 0; nt < 4; nt++)
        acc[mt][nt] = WMMA_F32_F16(af[mt].v, bf[nt].v, acc[mt][nt]);
  };

  const int NT = K >> 5;
  issue_tile(0, 0);
  int kt = 0;
  for (; kt < NT - 1; kt++) {
    issue_tile((kt + 1) & 1, (kt + 1) * 32);  // prefetch next tile (other buffer)
    WAIT_ASYNC(4);                            // wait for current tile only
    __syncthreads();
    compute_tile(kt & 1);
    __syncthreads();
  }
  WAIT_ASYNC(0);
  __syncthreads();
  compute_tile(kt & 1);

  #pragma unroll
  for (int mt = 0; mt < 2; mt++)
    #pragma unroll
    for (int nt = 0; nt < 4; nt++)
      #pragma unroll
      for (int v = 0; v < 8; v++) {
        int grow = bR + wm * 32 + mt * 16 + v + 8 * lhalf;
        int gcol = bC + wn * 64 + nt * 16 + l15;
        float val = acc[mt][nt][v];
        if constexpr (EPI == 0) {
          int c = gcol >> 10, hh = (gcol >> 6) & 15, d = gcol & 63;
          int bb = grow >> 10, n = grow & 1023;
          int bh = bb * 16 + hh;
          _Float16 hv = (_Float16)val;
          if (c == 0)      qout[((size_t)bh * 1024 + n) * 64 + d] = hv;
          else if (c == 1) kout[((size_t)bh * 1024 + n) * 64 + d] = hv;
          else             vout[((size_t)bh * 64 + d) * 1024 + n] = hv;  // V stored transposed
        } else if constexpr (EPI == 1) {
          size_t o = (size_t)grow * N + gcol;
          outF[o] = resid[o] + gamma[gcol] * (val + bias[gcol]);
        } else {
          float t = val + bias[gcol];
          outH[(size_t)grow * N + gcol] =
              (_Float16)(0.5f * t * (1.0f + erff(t * 0.70710678118654752f)));
        }
      }
}

// ---------------------------------------------------------------------------
// Flash attention: grid (16 row-tiles, 128 bh), 128 threads = 4 waves.
// Each wave owns 16 query rows; streams 64-key tiles (K rows + V^T rows in
// double-buffered LDS with async copies overlapping compute).
// Online softmax per row (permlane16 butterflies); P via per-wave LDS region.
// ---------------------------------------------------------------------------
__global__ __launch_bounds__(128) void attn_kernel(
    const _Float16* __restrict__ Q, const _Float16* __restrict__ K,
    const _Float16* __restrict__ V, _Float16* __restrict__ O) {
  __shared__ __align__(16) _Float16 Ks[2][64 * 88];
  __shared__ __align__(16) _Float16 VTs[2][64 * 88];
  __shared__ __align__(16) _Float16 Ps[4][16 * 88];
  const int tid = threadIdx.x, lane = tid & 31, wave = tid >> 5;
  const int lhalf = lane >> 4, l15 = lane & 15;
  const int bh = blockIdx.y, rt = blockIdx.x;
  const int r0 = rt * 64 + wave * 16;
  const _Float16* Qb = Q + ((size_t)bh * 1024 + r0) * 64;

  AFrag qf[2];
  #pragma unroll
  for (int s2 = 0; s2 < 2; s2++) {
    const _Float16* p = Qb + l15 * 64 + s2 * 32 + lhalf * 8;
    qf[s2].h8[0] = *(const v8h*)p;
    qf[s2].h8[1] = *(const v8h*)(p + 16);
    #pragma unroll
    for (int j = 0; j < 16; j++) qf[s2].h[j] = qf[s2].h[j] * (_Float16)0.125f;  // 1/sqrt(64)
  }

  v8f oacc[4] = {};
  float rmax[8], rsum[8];
  #pragma unroll
  for (int v = 0; v < 8; v++) { rmax[v] = -3.0e38f; rsum[v] = 0.f; }
  _Float16* P = &Ps[wave][0];

  auto issue_tile = [&](int buf, int kt) {
    #pragma unroll
    for (int i = 0; i < 4; i++) {
      int idx = tid + i * 128;
      int r = idx >> 3, c = (idx & 7) * 8;
      copy16_g2l(&K[((size_t)bh * 1024 + kt * 64 + r) * 64 + c], &Ks[buf][r * 88 + c]);
      copy16_g2l(&V[((size_t)bh * 64 + r) * 1024 + kt * 64 + c], &VTs[buf][r * 88 + c]);
    }
  };

  auto compute_tile = [&](int buf) {
    v8f sc[4] = {};
    #pragma unroll
    for (int s2 = 0; s2 < 2; s2++)
      #pragma unroll
      for (int nt = 0; nt < 4; nt++) {
        AFrag bf;
        const _Float16* p = &Ks[buf][(nt * 16 + l15) * 88 + s2 * 32 + lhalf * 8];
        bf.h8[0] = *(const v8h*)p; bf.h8[1] = *(const v8h*)(p + 16);
        sc[nt] = WMMA_F32_F16(qf[s2].v, bf.v, sc[nt]);
      }

    float fsc[8];
    #pragma unroll
    for (int v = 0; v < 8; v++) {
      float mn = fmaxf(fmaxf(sc[0][v], sc[1][v]), fmaxf(sc[2][v], sc[3][v]));
      mn = rowmax16(mn);
      mn = fmaxf(mn, rmax[v]);
      float ps = 0.f;
      #pragma unroll
      for (int nt = 0; nt < 4; nt++) { float p_ = __expf(sc[nt][v] - mn); sc[nt][v] = p_; ps += p_; }
      ps = rowsum16(ps);
      fsc[v] = __expf(rmax[v] - mn);
      rsum[v] = rsum[v] * fsc[v] + ps;
      rmax[v] = mn;
    }
    #pragma unroll
    for (int nt = 0; nt < 4; nt++)
      #pragma unroll
      for (int v = 0; v < 8; v++) oacc[nt][v] *= fsc[v];

    // C-layout P -> per-wave LDS (same-wave DS ops are in order: no barrier needed)
    #pragma unroll
    for (int nt = 0; nt < 4; nt++)
      #pragma unroll
      for (int v = 0; v < 8; v++)
        P[(v + 8 * lhalf) * 88 + nt * 16 + l15] = (_Float16)sc[nt][v];

    #pragma unroll
    for (int s2 = 0; s2 < 2; s2++) {
      AFrag pa;
      const _Float16* pp = &P[l15 * 88 + s2 * 32 + lhalf * 8];
      pa.h8[0] = *(const v8h*)pp; pa.h8[1] = *(const v8h*)(pp + 16);
      #pragma unroll
      for (int nt = 0; nt < 4; nt++) {
        AFrag bv;
        const _Float16* pv = &VTs[buf][(nt * 16 + l15) * 88 + s2 * 32 + lhalf * 8];
        bv.h8[0] = *(const v8h*)pv; bv.h8[1] = *(const v8h*)(pv + 16);
        oacc[nt] = WMMA_F32_F16(pa.v, bv.v, oacc[nt]);
      }
    }
  };

  issue_tile(0, 0);
  int kt = 0;
  for (; kt < 15; kt++) {
    issue_tile((kt + 1) & 1, kt + 1);  // prefetch next key tile (other buffer)
    WAIT_ASYNC(8);                     // wait for current tile's 8 copies only
    __syncthreads();
    compute_tile(kt & 1);
    __syncthreads();
  }
  WAIT_ASYNC(0);
  __syncthreads();
  compute_tile(kt & 1);

  const int b = bh >> 4, hh = bh & 15;
  #pragma unroll
  for (int v = 0; v < 8; v++) rsum[v] = 1.0f / rsum[v];
  #pragma unroll
  for (int nt = 0; nt < 4; nt++)
    #pragma unroll
    for (int v = 0; v < 8; v++) {
      int nrow = r0 + v + 8 * lhalf;
      int d = nt * 16 + l15;
      O[((size_t)b * 1024 + nrow) * 1024 + hh * 64 + d] = (_Float16)(oacc[nt][v] * rsum[v]);
    }
}

// ---------------------------------------------------------------------------
// Host launch
// ---------------------------------------------------------------------------
extern "C" void kernel_launch(void* const* d_in, const int* in_sizes, int n_in,
                              void* d_out, int out_size, void* d_ws, size_t ws_size,
                              hipStream_t stream) {
  (void)in_sizes; (void)n_in; (void)out_size; (void)ws_size;
  const float* x      = (const float*)d_in[0];
  const float* ln1_w  = (const float*)d_in[1];
  const float* ln1_b  = (const float*)d_in[2];
  const float* qkv_w  = (const float*)d_in[3];
  const float* proj_w = (const float*)d_in[4];
  const float* proj_b = (const float*)d_in[5];
  const float* ln2_w  = (const float*)d_in[6];
  const float* ln2_b  = (const float*)d_in[7];
  const float* fc1_w  = (const float*)d_in[8];
  const float* fc1_b  = (const float*)d_in[9];
  const float* fc2_w  = (const float*)d_in[10];
  const float* fc2_b  = (const float*)d_in[11];
  const float* gamma1 = (const float*)d_in[12];
  const float* gamma2 = (const float*)d_in[13];
  float* out = (float*)d_out;

  char* ws = (char*)d_ws;
  size_t off = 0;
  auto alloc = [&](size_t bytes) -> void* {
    void* p = ws + off;
    off += (bytes + 255) & ~(size_t)255;
    return p;
  };
  const size_t T = 8192;  // tokens
  _Float16* qkv_wh = (_Float16*)alloc(3072 * 1024 * 2);
  _Float16* proj_wh = (_Float16*)alloc(1024 * 1024 * 2);
  _Float16* fc1_wh = (_Float16*)alloc((size_t)4096 * 1024 * 2);
  _Float16* fc2_wh = (_Float16*)alloc((size_t)1024 * 4096 * 2);
  float*    x1     = (float*)alloc(T * 1024 * 4);
  _Float16* xh     = (_Float16*)alloc(T * 1024 * 2);       // reused as h2 (LN2 output)
  char*     regB   = (char*)alloc(4 * T * 1024 * 2);       // Q|K|V|attn, reused as h3
  _Float16* q      = (_Float16*)(regB);
  _Float16* k      = (_Float16*)(regB + 1 * T * 1024 * 2);
  _Float16* v      = (_Float16*)(regB + 2 * T * 1024 * 2);
  _Float16* attn_h = (_Float16*)(regB + 3 * T * 1024 * 2);
  _Float16* h3     = (_Float16*)(regB);                    // [8192 x 4096] after attn consumed

  auto cvt = [&](const float* src, _Float16* dst, int n) {
    int n4 = n / 4;
    cvt_f32_f16<<<(n4 + 255) / 256, 256, 0, stream>>>(src, dst, n4);
  };
  cvt(qkv_w, qkv_wh, 3072 * 1024);
  cvt(proj_w, proj_wh, 1024 * 1024);
  cvt(fc1_w, fc1_wh, 4096 * 1024);
  cvt(fc2_w, fc2_wh, 1024 * 4096);

  // LN1 -> f16
  ln_kernel<<<8192, 256, 0, stream>>>(x, ln1_w, ln1_b, xh);
  // QKV GEMM with scatter (V transposed)
  gemm128<0><<<dim3(24, 64), 256, 0, stream>>>(xh, qkv_wh, nullptr, nullptr, nullptr,
                                               nullptr, nullptr, q, k, v, 3072, 1024);
  // Flash attention
  attn_kernel<<<dim3(16, 128), 128, 0, stream>>>(q, k, v, attn_h);
  // proj + bias + LayerScale residual -> x1 (fp32)
  gemm128<1><<<dim3(8, 64), 256, 0, stream>>>(attn_h, proj_wh, proj_b, gamma1, x, x1,
                                              nullptr, nullptr, nullptr, nullptr, 1024, 1024);
  // LN2 -> f16 (reuse xh)
  ln_kernel<<<8192, 256, 0, stream>>>(x1, ln2_w, ln2_b, xh);
  // FC1 + exact GELU -> h3 (f16)
  gemm128<2><<<dim3(32, 64), 256, 0, stream>>>(xh, fc1_wh, fc1_b, nullptr, nullptr,
                                               nullptr, h3, nullptr, nullptr, nullptr, 4096, 1024);
  // FC2 + bias + LayerScale residual -> out (fp32)
  gemm128<1><<<dim3(8, 64), 256, 0, stream>>>(h3, fc2_wh, fc2_b, gamma2, x1, out,
                                              nullptr, nullptr, nullptr, nullptr, 1024, 4096);
}
